// Text2graph_69595650064834
// MI455X (gfx1250) — matrix operand order repaired and checked by reading, not compiled, not run on hardware
//
#include <hip/hip_runtime.h>

// ---------------------------------------------------------------------------
// Shapes (compile-time constants from the reference)
// ---------------------------------------------------------------------------
#define BB      4
#define HH      16
#define LQ      512
#define LK      2048
#define DM      1024
#define DHEAD   64

typedef __attribute__((ext_vector_type(16))) _Float16 v16h;
typedef __attribute__((ext_vector_type(8)))  _Float16 v8h;
typedef __attribute__((ext_vector_type(8)))  float    v8f;

// ---------------------------------------------------------------------------
// CDNA5 async global->LDS copy (ASYNCcnt-tracked DMA, 16B per lane)
// ---------------------------------------------------------------------------
__device__ __forceinline__ unsigned lds_off32(const void* p) {
    typedef __attribute__((address_space(3))) char lchar;
    return (unsigned)(unsigned long long)(lchar*)(void*)(const_cast<void*>(p));
}
__device__ __forceinline__ void async_ld_b128(unsigned lds_dst, const void* gsrc) {
    asm volatile("global_load_async_to_lds_b128 %0, %1, off"
                 :: "v"(lds_dst), "v"(gsrc) : "memory");
}
__device__ __forceinline__ void wait_async_le4() {
    asm volatile("s_wait_asynccnt 0x4" ::: "memory");
}
__device__ __forceinline__ void wait_async_0() {
    asm volatile("s_wait_asynccnt 0x0" ::: "memory");
}

// ---------------------------------------------------------------------------
// WMMA helpers (CDNA5 16x16x32 f16 -> f32)
// ---------------------------------------------------------------------------
__device__ __forceinline__ v16h cat16(v8h lo, v8h hi) {
    return __builtin_shufflevector(lo, hi, 0,1,2,3,4,5,6,7,8,9,10,11,12,13,14,15);
}

// A-matrix 16x32 f16 fragment from a row-major [.., ld] f16 source (global or LDS).
// Lane L(0-15): row L, K={0..7,16..23}; lane L+16: row L, K={8..15,24..31}.
__device__ __forceinline__ v16h load_afrag(const _Float16* base,
                                           int ld, int row0, int k0, int lane) {
    const int r   = lane & 15;
    const int sel = (lane >> 4) & 1;
    const _Float16* p = base + (size_t)(row0 + r) * ld + k0 + sel * 8;
    v8h c0 = *(const v8h*)(p);
    v8h c1 = *(const v8h*)(p + 16);
    return cat16(c0, c1);
}

// B-matrix 32x16 f16 fragment from an N-major [N][K] (ldT) source (global or LDS).
// Lane L(0-15): column N=n0+L, K={0..7,16..23}; lane L+16: K={8..15,24..31}.
__device__ __forceinline__ v16h load_bfrag(const _Float16* baseT,
                                           int ldT, int k0, int n0, int lane) {
    const int r   = lane & 15;
    const int sel = (lane >> 4) & 1;
    const _Float16* p = baseT + (size_t)(n0 + r) * ldT + k0 + sel * 8;
    v8h c0 = *(const v8h*)(p);
    v8h c1 = *(const v8h*)(p + 16);
    return cat16(c0, c1);
}

__device__ __forceinline__ v8f wmma_f16(v16h a, v16h b, v8f c) {
    // (neg_a, A, neg_b, B, c_mod, C, reuse_a, reuse_b)
    return __builtin_amdgcn_wmma_f32_16x16x32_f16(false, a, false, b,
                                                  (short)0, c, false, false);
}

// ---------------------------------------------------------------------------
// Conversion kernels
// ---------------------------------------------------------------------------
__global__ void k_cvt_f16(const float* __restrict__ in, _Float16* __restrict__ out, int n) {
    int i = blockIdx.x * blockDim.x + threadIdx.x;
    if (i < n) out[i] = (_Float16)in[i];
}

// W [K=in][N=out] f32  ->  WT [N][K] f16 (optionally scaled)
__global__ void k_cvt_wT(const float* __restrict__ W, _Float16* __restrict__ WT, float scale) {
    int i = blockIdx.x * blockDim.x + threadIdx.x;   // over 1024*1024
    int k = i >> 10;
    int n = i & 1023;
    WT[(size_t)n * DM + k] = (_Float16)(W[(size_t)k * DM + n] * scale);
}

// ---------------------------------------------------------------------------
// Projection GEMM: out = X[M,1024] @ W + b, written per-head f16.
// A panel (16 x 1024) staged once per block into LDS via async DMA and shared
// by all 4 waves; B-frags stream from global (L2-resident weights).
// LDS A row stride padded to 1032 halves -> conflict-free 16-lane frag reads.
// transposed==0 : out[b][h][l][d]   (q, k)
// transposed==1 : out[b][h][d][l]   (v, for contiguous B-frags in alpha@V)
// ---------------------------------------------------------------------------
#define ALD 1032

__global__ __launch_bounds__(128)
void k_gemm_qkv(const _Float16* __restrict__ X, const _Float16* __restrict__ WT,
                const float* __restrict__ bias, float bias_scale,
                _Float16* __restrict__ out, int Lseq, int transposed) {
    const int lane = threadIdx.x & 31;
    const int wave = threadIdx.x >> 5;
    const int m0 = blockIdx.x * 16;
    const int n0 = blockIdx.y * 256 + wave * 64;

    __shared__ __align__(16) _Float16 Al[16 * ALD];
    // stage A panel: 16 rows x 1024 halves = 2048 x 16B chunks, 16 per thread
#pragma unroll
    for (int i = 0; i < 16; ++i) {
        int c = threadIdx.x + i * 128;
        int row = c >> 7, part = c & 127;
        async_ld_b128(lds_off32(&Al[row * ALD + part * 8]),
                      X + (size_t)(m0 + row) * DM + part * 8);
    }
    wait_async_0();
    __syncthreads();

    v8f acc[4] = {};
    for (int k0 = 0; k0 < DM; k0 += 32) {
        v16h a = load_afrag(Al, ALD, 0, k0, lane);
        v16h b0 = load_bfrag(WT, DM, k0, n0,      lane);
        v16h b1 = load_bfrag(WT, DM, k0, n0 + 16, lane);
        v16h b2 = load_bfrag(WT, DM, k0, n0 + 32, lane);
        v16h b3 = load_bfrag(WT, DM, k0, n0 + 48, lane);
        acc[0] = wmma_f16(a, b0, acc[0]);
        acc[1] = wmma_f16(a, b1, acc[1]);
        acc[2] = wmma_f16(a, b2, acc[2]);
        acc[3] = wmma_f16(a, b3, acc[3]);
    }

    const int r16 = lane & 15, sel = (lane >> 4) & 1;
#pragma unroll
    for (int t = 0; t < 4; ++t) {
        const int n  = n0 + t * 16 + r16;
        const float bv = bias[n] * bias_scale;
        const int hh = n >> 6, d = n & 63;
        if (!transposed) {
#pragma unroll
            for (int r = 0; r < 8; ++r) {
                int row = m0 + r + 8 * sel;
                int b = row / Lseq, l = row % Lseq;
                out[((size_t)(b * HH + hh) * Lseq + l) * DHEAD + d] =
                    (_Float16)(acc[t][r] + bv);
            }
        } else {
            int row0 = m0 + 8 * sel;
            int b = row0 / Lseq, l0 = row0 % Lseq;
            v8h pk;
#pragma unroll
            for (int r = 0; r < 8; ++r) pk[r] = (_Float16)(acc[t][r] + bv);
            *(v8h*)(out + ((size_t)(b * HH + hh) * DHEAD + d) * Lseq + l0) = pk;
        }
    }
}

// ---------------------------------------------------------------------------
// Flash attention: 4 waves per block share one (b,h); each wave owns a 16-row
// q tile. K/V tiles (32 wide) double-buffered in LDS via async DMA so the next
// tile's copy overlaps the current tile's WMMA + softmax.
// Row strides: K tile 72 halves, V/P 40 halves -> conflict-free frag reads.
// ---------------------------------------------------------------------------
#define KLD 72
#define VLD 40

__global__ __launch_bounds__(128)
void k_attn(const _Float16* __restrict__ qh, const _Float16* __restrict__ kh,
            const _Float16* __restrict__ vT, const int* __restrict__ mask,
            _Float16* __restrict__ ctx) {
    const int QB = LQ / 64;                 // q-blocks (of 4 tiles) per (b,h)
    const int bid = blockIdx.x;
    const int qblk = bid % QB;
    const int h = (bid / QB) % HH;
    const int b = bid / (QB * HH);
    const int tid = threadIdx.x;
    const int wave = tid >> 5;
    const int lane = tid & 31;
    const int r16 = lane & 15, sel = (lane >> 4) & 1;
    const int qt = qblk * 4 + wave;         // this wave's 16-row q tile

    const _Float16* qb = qh + ((size_t)(b * HH + h) * LQ + qt * 16) * DHEAD;
    const _Float16* kb = kh + (size_t)(b * HH + h) * LK * DHEAD;
    const _Float16* vb = vT + (size_t)(b * HH + h) * DHEAD * LK;
    const int* mb = mask + b * LK;

    __shared__ __align__(16) _Float16 Kl[2][32 * KLD];   // [j][d], padded
    __shared__ __align__(16) _Float16 Vl[2][64 * VLD];   // [d][j], padded
    __shared__ __align__(16) _Float16 Pl[4][16 * VLD];   // per-wave P staging

    const v16h aq0 = load_afrag(qb, DHEAD, 0, 0,  lane);
    const v16h aq1 = load_afrag(qb, DHEAD, 0, 32, lane);

    v8f O[4] = {};
    float mrow[8], lrow[8];
#pragma unroll
    for (int r = 0; r < 8; ++r) { mrow[r] = -1e30f; lrow[r] = 0.f; }

    // cooperative async stage of one 32-wide K/V tile (4 instr/wave)
    auto stage = [&](int bufi, int j0s) {
#pragma unroll
        for (int i = 0; i < 2; ++i) {       // K: 32 rows x 8 chunks
            int c = tid + i * 128;
            int row = c >> 3, part = c & 7;
            async_ld_b128(lds_off32(&Kl[bufi][row * KLD + part * 8]),
                          kb + (size_t)(j0s + row) * DHEAD + part * 8);
        }
#pragma unroll
        for (int i = 0; i < 2; ++i) {       // V: 64 rows x 4 chunks
            int c = tid + i * 128;
            int row = c >> 2, part = c & 3;
            async_ld_b128(lds_off32(&Vl[bufi][row * VLD + part * 8]),
                          vb + (size_t)row * LK + j0s + part * 8);
        }
    };

    stage(0, 0);
    for (int it = 0; it < LK / 32; ++it) {
        const int j0 = it * 32;
        const int cur = it & 1;
        if (it + 1 < LK / 32) { stage(cur ^ 1, j0 + 32); wait_async_le4(); }
        else                  { wait_async_0(); }
        __syncthreads();                    // current tile visible to all waves

        // mask as additive bias baked into the WMMA C initializer
        const float mk0 = (mb[j0 + r16] == 0)      ? -1e30f : 0.0f;
        const float mk1 = (mb[j0 + 16 + r16] == 0) ? -1e30f : 0.0f;
        v8f S0 = {mk0, mk0, mk0, mk0, mk0, mk0, mk0, mk0};
        v8f S1 = {mk1, mk1, mk1, mk1, mk1, mk1, mk1, mk1};

        // S = q @ k^T   (B-source = LDS K tile [j][d], N-major for N=j)
        v16h bk00 = load_bfrag(Kl[cur], KLD, 0,  0,  lane);
        v16h bk01 = load_bfrag(Kl[cur], KLD, 32, 0,  lane);
        v16h bk10 = load_bfrag(Kl[cur], KLD, 0,  16, lane);
        v16h bk11 = load_bfrag(Kl[cur], KLD, 32, 16, lane);
        S0 = wmma_f16(aq0, bk00, S0);  S0 = wmma_f16(aq1, bk01, S0);
        S1 = wmma_f16(aq0, bk10, S1);  S1 = wmma_f16(aq1, bk11, S1);

        // online softmax; rows live as (vgpr r, lane-half), cols across 16 lanes
#pragma unroll
        for (int r = 0; r < 8; ++r) {
            float t = fmaxf(S0[r], S1[r]);
            t = fmaxf(t, __shfl_xor(t, 1));
            t = fmaxf(t, __shfl_xor(t, 2));
            t = fmaxf(t, __shfl_xor(t, 4));
            t = fmaxf(t, __shfl_xor(t, 8));
            float mn   = fmaxf(mrow[r], t);
            float corr = __expf(mrow[r] - mn);
            float p0 = __expf(S0[r] - mn);
            float p1 = __expf(S1[r] - mn);
            float ps = p0 + p1;
            ps += __shfl_xor(ps, 1);
            ps += __shfl_xor(ps, 2);
            ps += __shfl_xor(ps, 4);
            ps += __shfl_xor(ps, 8);
            lrow[r] = lrow[r] * corr + ps;
            mrow[r] = mn;
            O[0][r] *= corr; O[1][r] *= corr; O[2][r] *= corr; O[3][r] *= corr;
            const int row = r + 8 * sel;
            Pl[wave][row * VLD + r16]      = (_Float16)p0;
            Pl[wave][row * VLD + 16 + r16] = (_Float16)p1;
        }
        __builtin_amdgcn_wave_barrier();    // keep per-wave LDS store->load order

        // P (C layout) -> A layout via per-wave LDS staging
        v16h AP = load_afrag(Pl[wave], VLD, 0, 0, lane);
        __builtin_amdgcn_wave_barrier();

        // O += P @ Vtile   (B-source = LDS V tile [d][j], N-major for N=d)
        v16h bv0 = load_bfrag(Vl[cur], VLD, 0, 0,  lane);
        v16h bv1 = load_bfrag(Vl[cur], VLD, 0, 16, lane);
        v16h bv2 = load_bfrag(Vl[cur], VLD, 0, 32, lane);
        v16h bv3 = load_bfrag(Vl[cur], VLD, 0, 48, lane);
        O[0] = wmma_f16(AP, bv0, O[0]);
        O[1] = wmma_f16(AP, bv1, O[1]);
        O[2] = wmma_f16(AP, bv2, O[2]);
        O[3] = wmma_f16(AP, bv3, O[3]);

        __syncthreads();                    // all waves done before tile reuse
    }

    // normalize and store ctx[b][l][h*64+d] f16 (natural, A-ready for out proj)
    _Float16* cb = ctx + ((size_t)(b * LQ + qt * 16)) * DM + h * DHEAD;
#pragma unroll
    for (int r = 0; r < 8; ++r) {
        const float inv = 1.0f / lrow[r];
        const int row = r + 8 * sel;
#pragma unroll
        for (int t = 0; t < 4; ++t)
            cb[(size_t)row * DM + t * 16 + r16] = (_Float16)(O[t][r] * inv);
    }
}

// ---------------------------------------------------------------------------
// Output projection: y = ctx @ Wo + bo + Q   (f32 out, residual fused)
// Same async-staged A panel as k_gemm_qkv.
// ---------------------------------------------------------------------------
__global__ __launch_bounds__(128)
void k_gemm_out(const _Float16* __restrict__ ctx, const _Float16* __restrict__ WoT,
                const float* __restrict__ bo, const float* __restrict__ Qin,
                float* __restrict__ y) {
    const int lane = threadIdx.x & 31;
    const int wave = threadIdx.x >> 5;
    const int m0 = blockIdx.x * 16;
    const int n0 = blockIdx.y * 256 + wave * 64;

    __shared__ __align__(16) _Float16 Al[16 * ALD];
#pragma unroll
    for (int i = 0; i < 16; ++i) {
        int c = threadIdx.x + i * 128;
        int row = c >> 7, part = c & 127;
        async_ld_b128(lds_off32(&Al[row * ALD + part * 8]),
                      ctx + (size_t)(m0 + row) * DM + part * 8);
    }
    wait_async_0();
    __syncthreads();

    v8f acc[4] = {};
    for (int k0 = 0; k0 < DM; k0 += 32) {
        v16h a = load_afrag(Al, ALD, 0, k0, lane);
        v16h b0 = load_bfrag(WoT, DM, k0, n0,      lane);
        v16h b1 = load_bfrag(WoT, DM, k0, n0 + 16, lane);
        v16h b2 = load_bfrag(WoT, DM, k0, n0 + 32, lane);
        v16h b3 = load_bfrag(WoT, DM, k0, n0 + 48, lane);
        acc[0] = wmma_f16(a, b0, acc[0]);
        acc[1] = wmma_f16(a, b1, acc[1]);
        acc[2] = wmma_f16(a, b2, acc[2]);
        acc[3] = wmma_f16(a, b3, acc[3]);
    }

    const int r16 = lane & 15, sel = (lane >> 4) & 1;
#pragma unroll
    for (int t = 0; t < 4; ++t) {
        const int n = n0 + t * 16 + r16;
        const float bv = bo[n];
#pragma unroll
        for (int r = 0; r < 8; ++r) {
            const int row = m0 + r + 8 * sel;
            const size_t idx = (size_t)row * DM + n;
            y[idx] = acc[t][r] + bv + Qin[idx];
        }
    }
}

// ---------------------------------------------------------------------------
// LayerNorm over last dim (1024); one 256-thread block per row.
// ---------------------------------------------------------------------------
__global__ __launch_bounds__(256)
void k_layernorm(const float* __restrict__ y, const float* __restrict__ gamma,
                 const float* __restrict__ beta, float* __restrict__ out) {
    const int row = blockIdx.x;
    const int tid = threadIdx.x;
    const float* x = y + (size_t)row * DM;
    float s = 0.f, ss = 0.f;
    for (int i = tid; i < DM; i += 256) { float v = x[i]; s += v; ss += v * v; }
    __shared__ float S1[256], S2[256];
    S1[tid] = s; S2[tid] = ss;
    __syncthreads();
    for (int off = 128; off > 0; off >>= 1) {
        if (tid < off) { S1[tid] += S1[tid + off]; S2[tid] += S2[tid + off]; }
        __syncthreads();
    }
    const float mu  = S1[0] * (1.0f / DM);
    const float var = S2[0] * (1.0f / DM) - mu * mu;
    const float rs  = rsqrtf(var + 1e-5f);
    for (int i = tid; i < DM; i += 256)
        out[(size_t)row * DM + i] = (x[i] - mu) * rs * gamma[i] + beta[i];
}

// ---------------------------------------------------------------------------
// Host launcher
// ---------------------------------------------------------------------------
extern "C" void kernel_launch(void* const* d_in, const int* in_sizes, int n_in,
                              void* d_out, int out_size, void* d_ws, size_t ws_size,
                              hipStream_t stream) {
    (void)in_sizes; (void)n_in; (void)out_size; (void)ws_size;
    const float* Q  = (const float*)d_in[0];
    const float* K  = (const float*)d_in[1];
    const float* V  = (const float*)d_in[2];
    const int* mask = (const int*)d_in[4];
    const float* Wq = (const float*)d_in[5];
    const float* bq = (const float*)d_in[6];
    const float* Wk = (const float*)d_in[7];
    const float* bk = (const float*)d_in[8];
    const float* Wv = (const float*)d_in[9];
    const float* bv = (const float*)d_in[10];
    const float* Wo = (const float*)d_in[11];
    const float* bo = (const float*)d_in[12];
    const float* gamma = (const float*)d_in[13];
    const float* beta  = (const float*)d_in[14];

    const size_t nQ = (size_t)BB * LQ * DM;   // 2,097,152
    const size_t nK = (size_t)BB * LK * DM;   // 8,388,608
    const size_t nW = (size_t)DM * DM;        // 1,048,576

    char* p = (char*)d_ws;
    auto alloc = [&](size_t bytes) -> void* {
        void* r = (void*)p;
        p += (bytes + 255) & ~(size_t)255;
        return r;
    };
    _Float16* Qh  = (_Float16*)alloc(nQ * 2);
    _Float16* Kh  = (_Float16*)alloc(nK * 2);
    _Float16* Vh  = (_Float16*)alloc(nK * 2);
    _Float16* WqT = (_Float16*)alloc(nW * 2);
    _Float16* WkT = (_Float16*)alloc(nW * 2);
    _Float16* WvT = (_Float16*)alloc(nW * 2);
    _Float16* WoT = (_Float16*)alloc(nW * 2);
    _Float16* qh  = (_Float16*)alloc(nQ * 2);
    _Float16* kh  = (_Float16*)alloc(nK * 2);
    _Float16* vTh = (_Float16*)alloc(nK * 2);
    _Float16* ctx = (_Float16*)alloc(nQ * 2);
    float*    yws = (float*)   alloc(nQ * 4);

    // 1) f32 -> f16 conversions (+ weight transpose; 1/sqrt(dk) folded into Wq/bq)
    k_cvt_f16<<<(int)((nQ + 255) / 256), 256, 0, stream>>>(Q, Qh, (int)nQ);
    k_cvt_f16<<<(int)((nK + 255) / 256), 256, 0, stream>>>(K, Kh, (int)nK);
    k_cvt_f16<<<(int)((nK + 255) / 256), 256, 0, stream>>>(V, Vh, (int)nK);
    k_cvt_wT<<<(int)(nW / 256), 256, 0, stream>>>(Wq, WqT, 0.125f);
    k_cvt_wT<<<(int)(nW / 256), 256, 0, stream>>>(Wk, WkT, 1.0f);
    k_cvt_wT<<<(int)(nW / 256), 256, 0, stream>>>(Wv, WvT, 1.0f);
    k_cvt_wT<<<(int)(nW / 256), 256, 0, stream>>>(Wo, WoT, 1.0f);

    // 2) projections (WMMA GEMMs, async-staged A panels)
    k_gemm_qkv<<<dim3(BB * LQ / 16, DM / 256), 128, 0, stream>>>(Qh, WqT, bq, 0.125f, qh,  LQ, 0);
    k_gemm_qkv<<<dim3(BB * LK / 16, DM / 256), 128, 0, stream>>>(Kh, WkT, bk, 1.0f,   kh,  LK, 0);
    k_gemm_qkv<<<dim3(BB * LK / 16, DM / 256), 128, 0, stream>>>(Vh, WvT, bv, 1.0f,   vTh, LK, 1);

    // 3) flash attention (WMMA + double-buffered async K/V tiles)
    k_attn<<<BB * HH * (LQ / 64), 128, 0, stream>>>(qh, kh, vTh, mask, ctx);

    // 4) output projection + residual (WMMA), then layernorm
    k_gemm_out<<<dim3(BB * LQ / 16, DM / 256), 128, 0, stream>>>(ctx, WoT, bo, Q, yws);
    k_layernorm<<<BB * LQ, 256, 0, stream>>>(yws, gamma, beta, (float*)d_out);
}
